// CyclicAttention_17892833755561
// MI455X (gfx1250) — matrix operand order repaired and checked by reading, not compiled, not run on hardware
//
#include <hip/hip_runtime.h>
#include <hip/hip_bf16.h>
#include <type_traits>

typedef __attribute__((ext_vector_type(16))) _Float16 v16h;
typedef __attribute__((ext_vector_type(8)))  _Float16 v8h;
typedef __attribute__((ext_vector_type(4)))  _Float16 v4h;
typedef __attribute__((ext_vector_type(8)))  float    v8f;
typedef __attribute__((ext_vector_type(4)))  float    v4f;

static __device__ __forceinline__ v8f zero8f() {
  v8f z;
#pragma unroll
  for (int i = 0; i < 8; ++i) z[i] = 0.0f;
  return z;
}

static __device__ __forceinline__ v16h cat16(v8h lo, v8h hi) {
  return __builtin_shufflevector(lo, hi, 0, 1, 2, 3, 4, 5, 6, 7, 8, 9, 10, 11, 12, 13, 14, 15);
}

static __device__ __forceinline__ v8f wmma_f16(v16h a, v16h b, v8f c) {
  // D = A(16x32 f16) * B(32x16 f16) + C(16x16 f32)
  return __builtin_amdgcn_wmma_f32_16x16x32_f16(false, a, false, b, (short)0, c, false, false);
}

// ---- async global->LDS copies (gfx1250 GLOBAL_LOAD_ASYNC_TO_LDS_B128) ----
#if __has_builtin(__builtin_amdgcn_global_load_async_to_lds_b128)
#define HAVE_ASYNC_LDS 1
#define AS1 __attribute__((address_space(1)))
#define AS3 __attribute__((address_space(3)))
typedef int b128_t __attribute__((vector_size(16)));
static __device__ __forceinline__ void async_copy16(const _Float16* g, _Float16* l) {
  __builtin_amdgcn_global_load_async_to_lds_b128((AS1 b128_t*)(g), (AS3 b128_t*)(l), 0, 0);
}
#if __has_builtin(__builtin_amdgcn_s_wait_asynccnt)
#define WAIT_ASYNC() __builtin_amdgcn_s_wait_asynccnt(0)
#else
#define WAIT_ASYNC() asm volatile("s_wait_asynccnt 0x0" ::: "memory")
#endif
#endif

// ---- xor-shuffle within 16-lane halves: v_permlane16_b32 (VALU, no LDS pipe) ----
#if __has_builtin(__builtin_amdgcn_permlane16)
template <unsigned LO, unsigned HI>
static __device__ __forceinline__ float permlane_xor(float x) {
  unsigned u = __builtin_bit_cast(unsigned, x);
  u = __builtin_amdgcn_permlane16(u, u, LO, HI, false, false);
  return __builtin_bit_cast(float, u);
}
static __device__ __forceinline__ float red_max16(float x) {
  x = fmaxf(x, permlane_xor<0x67452301u, 0xEFCDAB89u>(x));  // xor 1
  x = fmaxf(x, permlane_xor<0x54761032u, 0xDCFE98BAu>(x));  // xor 2
  x = fmaxf(x, permlane_xor<0x32107654u, 0xBA98FEDCu>(x));  // xor 4
  x = fmaxf(x, permlane_xor<0xFEDCBA98u, 0x76543210u>(x));  // xor 8
  return x;
}
static __device__ __forceinline__ float red_sum16(float x) {
  x += permlane_xor<0x67452301u, 0xEFCDAB89u>(x);
  x += permlane_xor<0x54761032u, 0xDCFE98BAu>(x);
  x += permlane_xor<0x32107654u, 0xBA98FEDCu>(x);
  x += permlane_xor<0xFEDCBA98u, 0x76543210u>(x);
  return x;
}
#else
static __device__ __forceinline__ float red_max16(float x) {
#pragma unroll
  for (int m = 1; m < 16; m <<= 1) x = fmaxf(x, __shfl_xor(x, m, 32));
  return x;
}
static __device__ __forceinline__ float red_sum16(float x) {
#pragma unroll
  for (int m = 1; m < 16; m <<= 1) x += __shfl_xor(x, m, 32);
  return x;
}
#endif

#if __has_builtin(__builtin_amdgcn_exp2f)
#define FAST_EXP2(x) __builtin_amdgcn_exp2f(x)
#else
#define FAST_EXP2(x) exp2f(x)
#endif

// ---------------------------------------------------------------------------
// GEMM: out[M,N] = A[M,K] @ W[K,N] + bias[N]; f16 WMMA with f32 accum.
// 128x128 block tile, BK=64, 256 threads = 8 waves (4M x 2N), each wave
// computes 32x64 (2x4 WMMA accs). LDS double-buffered + register pipelined.
// ---------------------------------------------------------------------------
template <typename TA, bool F16OUT>
__global__ __launch_bounds__(256) void gemm_bias_kernel(
    const TA* __restrict__ A, const float* __restrict__ W,
    const float* __restrict__ bias, void* __restrict__ outv,
    int M, int N, int K) {
  constexpr int BM = 128, BN = 128, BK = 64;
  constexpr int LDA = BK + 8;  // 72 halves -> 144B stride (16B aligned)
  constexpr int LDB = BK + 8;  // sB transposed: [n][k]

  __shared__ alignas(16) _Float16 sA[2 * BM * LDA];
  __shared__ alignas(16) _Float16 sB[2 * BN * LDB];

  const int tid  = threadIdx.x;
  const int lane = tid & 31;
  const int wave = tid >> 5;
  const int wm   = wave & 3;
  const int wn   = wave >> 2;
  const int l16  = lane & 15;
  const int kb   = lane >> 4;

  const int mBase = blockIdx.y * BM;
  const int nBase = blockIdx.x * BN;

  using VecA = typename std::conditional<sizeof(TA) == 4, v4f, v4h>::type;
  VecA ra[8];
  v4f  rb[8];

  v8f acc[2][4];
#pragma unroll
  for (int i = 0; i < 2; ++i)
#pragma unroll
    for (int j = 0; j < 4; ++j) acc[i][j] = zero8f();

  auto issueLoads = [&](int k0) {
#pragma unroll
    for (int it = 0; it < 8; ++it) {
      int i = tid + it * 256, row = i >> 4, col = (i & 15) * 4;
      ra[it] = *(const VecA*)(A + (size_t)(mBase + row) * K + k0 + col);
    }
#pragma unroll
    for (int it = 0; it < 8; ++it) {
      int i = tid + it * 256, kr = i >> 5, nc = (i & 31) * 4;
      rb[it] = *(const v4f*)(W + (size_t)(k0 + kr) * N + nBase + nc);
    }
  };

  auto storeLDS = [&](int buf) {
    _Float16* bA = sA + buf * (BM * LDA);
    _Float16* bB = sB + buf * (BN * LDB);
#pragma unroll
    for (int it = 0; it < 8; ++it) {
      int i = tid + it * 256, row = i >> 4, col = (i & 15) * 4;
      v4h hv;
      if constexpr (sizeof(TA) == 4) {
#pragma unroll
        for (int e = 0; e < 4; ++e) hv[e] = (_Float16)ra[it][e];
      } else {
        hv = ra[it];
      }
      *(v4h*)&bA[row * LDA + col] = hv;
    }
#pragma unroll
    for (int it = 0; it < 8; ++it) {
      int i = tid + it * 256, kr = i >> 5, nc = (i & 31) * 4;
#pragma unroll
      for (int e = 0; e < 4; ++e) bB[(nc + e) * LDB + kr] = (_Float16)rb[it][e];
    }
  };

  auto computeTile = [&](int buf) {
    const _Float16* bA = sA + buf * (BM * LDA);
    const _Float16* bB = sB + buf * (BN * LDB);
#pragma unroll
    for (int kc = 0; kc < BK / 32; ++kc) {
      v16h af[2];
#pragma unroll
      for (int mt = 0; mt < 2; ++mt) {
        const _Float16* p = &bA[(wm * 32 + mt * 16 + l16) * LDA + kc * 32 + kb * 8];
        af[mt] = cat16(*(const v8h*)p, *(const v8h*)(p + 16));
      }
      v16h bf[4];
#pragma unroll
      for (int nt = 0; nt < 4; ++nt) {
        const _Float16* p = &bB[(wn * 64 + nt * 16 + l16) * LDB + kc * 32 + kb * 8];
        bf[nt] = cat16(*(const v8h*)p, *(const v8h*)(p + 16));
      }
#pragma unroll
      for (int mt = 0; mt < 2; ++mt)
#pragma unroll
        for (int nt = 0; nt < 4; ++nt)
          acc[mt][nt] = wmma_f16(af[mt], bf[nt], acc[mt][nt]);
    }
  };

  // pipeline prologue
  issueLoads(0);
  storeLDS(0);
  __syncthreads();

  int buf = 0;
  for (int k0 = 0; k0 < K; k0 += BK) {
    const bool hasNext = (k0 + BK) < K;
    if (hasNext) {
      issueLoads(k0 + BK);  // global loads in flight while computing
      __builtin_prefetch(A + (size_t)(mBase + (tid >> 1)) * K + (k0 + 2 * BK) + (tid & 1) * 32, 0, 0);
    }
    computeTile(buf);
    if (hasNext) {
      storeLDS(buf ^ 1);
      __syncthreads();
    }
    buf ^= 1;
  }

  // epilogue: C/D layout lane(l16)=col, VGPR r -> row r + kb*8
#pragma unroll
  for (int mt = 0; mt < 2; ++mt)
#pragma unroll
    for (int nt = 0; nt < 4; ++nt) {
      const int n = nBase + wn * 64 + nt * 16 + l16;
      const float bv = bias[n];
      const int mrow = mBase + wm * 32 + mt * 16 + kb * 8;
      if constexpr (F16OUT) {
        _Float16* o = (_Float16*)outv;
#pragma unroll
        for (int r = 0; r < 8; ++r)
          o[(size_t)(mrow + r) * N + n] = (_Float16)(acc[mt][nt][r] + bv);
      } else {
        float* o = (float*)outv;
#pragma unroll
        for (int r = 0; r < 8; ++r)
          o[(size_t)(mrow + r) * N + n] = acc[mt][nt][r] + bv;
      }
    }
}

// ---------------------------------------------------------------------------
// Flash attention, one head per blockIdx.y, 64 query rows per block (4 waves
// x 16 rows). Streams 64-key KV tiles: 8 QK^T WMMAs + 8 PV WMMAs per tile,
// one online-softmax update per 64 keys (log2-domain, permlane reductions).
// Q/K tiles staged with gfx1250 async global->LDS copies when available.
// ---------------------------------------------------------------------------
__global__ __launch_bounds__(128) void flash_attn_kernel(
    const _Float16* __restrict__ Q, const _Float16* __restrict__ Km,
    const _Float16* __restrict__ Vm, _Float16* __restrict__ Om) {
  constexpr int S = 2048, H = 1024, NH = 16;
  constexpr int LQ = 72, LK = 72, LV = 72, LP = 72;

  __shared__ alignas(16) _Float16 sQ[64 * LQ];      // [qrow][d]
  __shared__ alignas(16) _Float16 sK[64 * LK];      // [key][d]
  __shared__ alignas(16) _Float16 sVt[64 * LV];     // [d][key] (transposed)
  __shared__ alignas(16) _Float16 sP[4 * 16 * LP];  // per-wave probs 16x64

  const int tid  = threadIdx.x;
  const int lane = tid & 31;
  const int wave = tid >> 5;
  const int l16  = lane & 15;
  const int kb   = lane >> 4;

  const int bh = blockIdx.y;
  const int b  = bh / NH;
  const int h  = bh % NH;
  const int qBase = blockIdx.x * 64;
  const size_t rowOff = (size_t)b * S * H + (size_t)h * 64;

  // stage 64x64 Q tile once (async DMA when available)
#pragma unroll
  for (int it = 0; it < 4; ++it) {
    int i = (tid + it * 128) * 8;
    int r = i >> 6, c = i & 63;
    const _Float16* gp = Q + rowOff + (size_t)(qBase + r) * H + c;
#if defined(HAVE_ASYNC_LDS)
    async_copy16(gp, &sQ[r * LQ + c]);
#else
    *(v8h*)&sQ[r * LQ + c] = *(const v8h*)gp;
#endif
  }
#if defined(HAVE_ASYNC_LDS)
  WAIT_ASYNC();
#endif
  __syncthreads();

  float mrow[8], lrow[8];
  v8f accO[4];
#pragma unroll
  for (int r = 0; r < 8; ++r) { mrow[r] = -1e30f; lrow[r] = 0.0f; }
#pragma unroll
  for (int dt = 0; dt < 4; ++dt) accO[dt] = zero8f();

  const float scl2 = 0.125f * 1.4426950408889634f;  // 1/sqrt(DK) * log2(e)

  for (int j0 = 0; j0 < S; j0 += 64) {
    // stage K tile 64x64 row-major (async DMA when available)
#pragma unroll
    for (int it = 0; it < 4; ++it) {
      int i = (tid + it * 128) * 8;
      int r = i >> 6, c = i & 63;
      const _Float16* gp = Km + rowOff + (size_t)(j0 + r) * H + c;
#if defined(HAVE_ASYNC_LDS)
      async_copy16(gp, &sK[r * LK + c]);
#else
      *(v8h*)&sK[r * LK + c] = *(const v8h*)gp;
#endif
    }
    // stage V tile 64x64 transposed into sVt[d][key] (needs reshape: manual)
    {
      int key = tid >> 1;
      int dbase = (tid & 1) * 32;
      const _Float16* vp = Vm + rowOff + (size_t)(j0 + key) * H + dbase;
      v8h a0 = *(const v8h*)vp;
      v8h a1 = *(const v8h*)(vp + 8);
      v8h a2 = *(const v8h*)(vp + 16);
      v8h a3 = *(const v8h*)(vp + 24);
#pragma unroll
      for (int e = 0; e < 8; ++e) {
        sVt[(dbase + e) * LV + key]      = a0[e];
        sVt[(dbase + 8 + e) * LV + key]  = a1[e];
        sVt[(dbase + 16 + e) * LV + key] = a2[e];
        sVt[(dbase + 24 + e) * LV + key] = a3[e];
      }
    }
    if (j0 + 64 < S) {  // prefetch next KV tile (global_prefetch_b8)
      __builtin_prefetch(Km + rowOff + (size_t)(j0 + 64 + (tid >> 1)) * H + (tid & 1) * 32, 0, 0);
      __builtin_prefetch(Vm + rowOff + (size_t)(j0 + 64 + (tid >> 1)) * H + (tid & 1) * 32, 0, 0);
    }
#if defined(HAVE_ASYNC_LDS)
    WAIT_ASYNC();
#endif
    __syncthreads();

    // ---- scores: four 16-key sub-tiles, d = 64 (2 WMMA each) ----
    v8f sc[4];
#pragma unroll
    for (int kt = 0; kt < 4; ++kt) {
      v8f a = zero8f();
#pragma unroll
      for (int kc = 0; kc < 2; ++kc) {
        const _Float16* ap = &sQ[(wave * 16 + l16) * LQ + kc * 32 + kb * 8];
        v16h af = cat16(*(const v8h*)ap, *(const v8h*)(ap + 16));
        const _Float16* bp = &sK[(kt * 16 + l16) * LK + kc * 32 + kb * 8];
        v16h bfv = cat16(*(const v8h*)bp, *(const v8h*)(bp + 16));
        a = wmma_f16(af, bfv, a);
      }
#pragma unroll
      for (int r = 0; r < 8; ++r) a[r] *= scl2;  // into log2 domain
      sc[kt] = a;
    }

    // ---- online softmax over this 64-key tile ----
    float tmax[8];
#pragma unroll
    for (int r = 0; r < 8; ++r)
      tmax[r] = red_max16(fmaxf(fmaxf(sc[0][r], sc[1][r]), fmaxf(sc[2][r], sc[3][r])));

    float corr[8], rsum[8];
#pragma unroll
    for (int r = 0; r < 8; ++r) {
      float mnew = fmaxf(mrow[r], tmax[r]);
      corr[r] = FAST_EXP2(mrow[r] - mnew);
      mrow[r] = mnew;
      _Float16* pw = &sP[(wave * 16 + (kb * 8 + r)) * LP];
      float rs = 0.0f;
#pragma unroll
      for (int kt = 0; kt < 4; ++kt) {
        float p = FAST_EXP2(sc[kt][r] - mnew);
        rs += p;
        pw[kt * 16 + l16] = (_Float16)p;
      }
      rsum[r] = rs;
    }
#pragma unroll
    for (int r = 0; r < 8; ++r) {
      rsum[r] = red_sum16(rsum[r]);
      lrow[r] = lrow[r] * corr[r] + rsum[r];
    }
#pragma unroll
    for (int dt = 0; dt < 4; ++dt)
#pragma unroll
      for (int r = 0; r < 8; ++r) accO[dt][r] *= corr[r];

    // sP is per-wave; LDS ops are in-order in HW; stop compiler reordering
    __builtin_amdgcn_wave_barrier();
    asm volatile("" ::: "memory");

    // ---- O += P @ V : 2 key-chunks x 4 d-tiles ----
#pragma unroll
    for (int kc2 = 0; kc2 < 2; ++kc2) {
      const _Float16* pp = &sP[(wave * 16 + l16) * LP + kc2 * 32 + kb * 8];
      v16h pf = cat16(*(const v8h*)pp, *(const v8h*)(pp + 16));
#pragma unroll
      for (int dt = 0; dt < 4; ++dt) {
        const _Float16* vp2 = &sVt[(dt * 16 + l16) * LV + kc2 * 32 + kb * 8];
        v16h vf = cat16(*(const v8h*)vp2, *(const v8h*)(vp2 + 16));
        accO[dt] = wmma_f16(pf, vf, accO[dt]);
      }
    }
    __syncthreads();
  }

  // ---- normalize and store merged (f16) ----
  float inv[8];
#pragma unroll
  for (int r = 0; r < 8; ++r) inv[r] = 1.0f / lrow[r];
#pragma unroll
  for (int dt = 0; dt < 4; ++dt)
#pragma unroll
    for (int r = 0; r < 8; ++r) {
      int qr = qBase + wave * 16 + kb * 8 + r;
      Om[rowOff + (size_t)qr * H + dt * 16 + l16] = (_Float16)(accO[dt][r] * inv[r]);
    }
}

// ---------------------------------------------------------------------------
extern "C" void kernel_launch(void* const* d_in, const int* in_sizes, int n_in,
                              void* d_out, int out_size, void* d_ws, size_t ws_size,
                              hipStream_t stream) {
  const float* q  = (const float*)d_in[0];
  const float* k  = (const float*)d_in[1];
  const float* v  = (const float*)d_in[2];
  const float* Wq = (const float*)d_in[3];
  const float* bq = (const float*)d_in[4];
  const float* Wk = (const float*)d_in[5];
  const float* bk = (const float*)d_in[6];
  const float* Wv = (const float*)d_in[7];
  const float* bv = (const float*)d_in[8];
  const float* Wo = (const float*)d_in[9];
  const float* bo = (const float*)d_in[10];
  const float* Wc = (const float*)d_in[11];
  const float* bc = (const float*)d_in[12];

  const int M = 2 * 2048;  // B*S
  const int H = 1024;
  const int C = 1024;

  _Float16* Qh = (_Float16*)d_ws;
  _Float16* Kh = Qh + (size_t)M * H;
  _Float16* Vh = Kh + (size_t)M * H;
  _Float16* Mg = Vh + (size_t)M * H;  // merged attention output

  float* out_c = (float*)d_out;  // tuple order: (c, h)
  float* out_h = out_c + (size_t)M * C;

  dim3 blk(256);
  dim3 grd(H / 128, M / 128);

  gemm_bias_kernel<float, true><<<grd, blk, 0, stream>>>(q, Wq, bq, Qh, M, H, H);
  gemm_bias_kernel<float, true><<<grd, blk, 0, stream>>>(k, Wk, bk, Kh, M, H, H);
  gemm_bias_kernel<float, true><<<grd, blk, 0, stream>>>(v, Wv, bv, Vh, M, H, H);

  dim3 agrd(2048 / 64, 2 * 16);  // (S/64, B*NH)
  flash_attn_kernel<<<agrd, dim3(128), 0, stream>>>(Qh, Kh, Vh, Mg);

  gemm_bias_kernel<_Float16, false><<<grd, blk, 0, stream>>>(Mg, Wc, bc, out_c, M, C, H);
  gemm_bias_kernel<_Float16, false><<<grd, blk, 0, stream>>>(Mg, Wo, bo, out_h, M, H, H);
}
